// Graph_Net_85624468013584
// MI455X (gfx1250) — matrix operand order repaired and checked by reading, not compiled, not run on hardware
//
#include <hip/hip_runtime.h>
#include <hip/hip_bf16.h>
#include <cstdint>

typedef __attribute__((ext_vector_type(16))) _Float16 v16h;
typedef __attribute__((ext_vector_type(8)))  _Float16 v8h;
typedef __attribute__((ext_vector_type(8)))  float    v8f;

#define N_NODES 20000
#define N_EDGES 320000
#define E_TOT   340000   /* 320000 edges + 20000 self loops */
#define N_TRAIN 100000
#define NEG_SLOPE 0.2f
#define GAT_EPS 1e-16f

// ---------------------------------------------------------------------------
// WMMA helpers (CDNA5 wave32, 16x16x32 f16 -> f32)
// ---------------------------------------------------------------------------
__device__ __forceinline__ v8f wmma16(v16h a, v16h b, v8f c) {
  return __builtin_amdgcn_wmma_f32_16x16x32_f16(false, a, false, b, (short)0, c,
                                                false, false);
}

// A fragment (16x32 f16, MxK), row-major LDS, lda multiple of 16 halfs.
// lanes 0-15: row=lane, K [0..15]; lanes 16-31: row=lane-16,
// K [16..31].  Two 16-byte contiguous runs -> 2x ds_load_b128.
__device__ __forceinline__ v16h load_a_frag(const _Float16* As, int lda,
                                            int row0, int k0) {
  int l = threadIdx.x & 31;
  const _Float16* p = As + (row0 + (l & 15)) * lda + k0 + ((l >> 4) << 4);
  v8h lo = *(const v8h*)p;
  v8h hi = *(const v8h*)(p + 8);
  return __builtin_shufflevector(lo, hi, 0, 1, 2, 3, 4, 5, 6, 7, 8, 9, 10, 11,
                                 12, 13, 14, 15);
}

// B fragment from swizzled storage.  Swizzle layout for a KxN f16 matrix
// (Kpad mult of 32, Npad mult of 16):
//   elem(k, n) at  ((k/32)*2 + (k%32)/16) * Npad*16  +  n*16  +  (k%16)
// => per-lane fragment (16 halfs) is 32 contiguous bytes -> 2x ds_load_b128.
__device__ __forceinline__ v16h load_b_swz(const _Float16* Bs, int npad,
                                           int kchunk, int n0) {
  int l = threadIdx.x & 31;
  const _Float16* p =
      Bs + ((size_t)((kchunk << 1) + (l >> 4)) * npad + n0 + (l & 15)) * 16;
  v8h lo = *(const v8h*)p;
  v8h hi = *(const v8h*)(p + 8);
  return __builtin_shufflevector(lo, hi, 0, 1, 2, 3, 4, 5, 6, 7, 8, 9, 10, 11,
                                 12, 13, 14, 15);
}

// C/D f32 16x16: vgpr j -> row j + 8*(lane>=16), col = lane%16
__device__ __forceinline__ void store_tile_f16(_Float16* out, int ldo, int row0,
                                               int n0, v8f acc,
                                               const float* bias, bool relu_) {
  int l = threadIdx.x & 31;
  int col = n0 + (l & 15);
  int rbase = row0 + ((l >> 4) << 3);
  float bv = bias[col];
#pragma unroll
  for (int j = 0; j < 8; ++j) {
    float v = acc[j] + bv;
    if (relu_) v = fmaxf(v, 0.0f);
    out[(rbase + j) * ldo + col] = (_Float16)v;
  }
}

__device__ __forceinline__ void atomicMaxF(float* addr, float val) {
  unsigned int* u = (unsigned int*)addr;
  unsigned int old = *u;
  while (__uint_as_float(old) < val) {
    unsigned int assumed = old;
    old = atomicCAS(u, assumed, __float_as_uint(val));
    if (old == assumed) break;
  }
}

// ---------------------------------------------------------------------------
// Fused point-MLP: relu(relu(relu(x@W1+b1)@W2+b2)@W3+b3).max(points)
// Block = 2 nodes = 128 rows, 8 waves; wave w owns a 16-row strip through all
// three layers.  Weights pre-swizzled into WMMA-B layout in global memory.
// ---------------------------------------------------------------------------
__global__ __launch_bounds__(256) void point_mlp_kernel(
    const float* __restrict__ x, const _Float16* __restrict__ w1s_g,
    const _Float16* __restrict__ w2s_g, const _Float16* __restrict__ w3s_g,
    const float* __restrict__ b1, const float* __restrict__ b2,
    const float* __restrict__ b3, float* __restrict__ feat) {
  __shared__ alignas(16) _Float16 a0[128 * 32];    // pts, K padded 16->32
  __shared__ alignas(16) _Float16 a1[128 * 64];    // layer1 out
  __shared__ alignas(16) _Float16 a2[128 * 128];   // layer2 out
  __shared__ alignas(16) _Float16 w1s[32 * 64];    // swizzled
  __shared__ alignas(16) _Float16 w2s[64 * 128];   // swizzled
  __shared__ alignas(16) _Float16 w3s[128 * 256];  // swizzled
  __shared__ float featbuf[2 * 256];

  const int t = threadIdx.x;
  const int node0 = blockIdx.x * 2;

  // stage pts: thread pair per row, each half-thread converts 8 floats
  {
    int r = t >> 1, c0 = (t & 1) * 8;
    const float* xp = x + (node0 * 64 + r) * 16 + c0;
    _Float16* dp = a0 + r * 32 + c0;
#pragma unroll
    for (int i = 0; i < 8; ++i) dp[i] = (_Float16)xp[i];
#pragma unroll
    for (int i = 0; i < 8; ++i) dp[16 + i] = (_Float16)0.0f;  // K pad 16->32
  }
  // stage swizzled weights: contiguous vector copies
  for (int i = t * 8; i < 32 * 64; i += 256 * 8)
    *(v8h*)(w1s + i) = *(const v8h*)(w1s_g + i);
  for (int i = t * 8; i < 64 * 128; i += 256 * 8)
    *(v8h*)(w2s + i) = *(const v8h*)(w2s_g + i);
  for (int i = t * 8; i < 128 * 256; i += 256 * 8)
    *(v8h*)(w3s + i) = *(const v8h*)(w3s_g + i);
  featbuf[t] = -3.402823466e+38f;
  featbuf[256 + t] = -3.402823466e+38f;
  __syncthreads();

  const int w = t >> 5;      // wave id == row tile
  const int lane = t & 31;
  const int row0 = w * 16;

  // ---- layer 1: 128x32 @ 32x64 ----
  {
    v16h a = load_a_frag(a0, 32, row0, 0);
#pragma unroll
    for (int c = 0; c < 4; ++c) {
      v16h b = load_b_swz(w1s, 64, 0, c * 16);
      v8f acc = {0.f, 0.f, 0.f, 0.f, 0.f, 0.f, 0.f, 0.f};
      acc = wmma16(a, b, acc);
      store_tile_f16(a1, 64, row0, c * 16, acc, b1, true);
    }
  }
  // ---- layer 2: 128x64 @ 64x128 ----
  {
    v16h a[2];
    a[0] = load_a_frag(a1, 64, row0, 0);
    a[1] = load_a_frag(a1, 64, row0, 32);
#pragma unroll
    for (int c = 0; c < 8; ++c) {
      v8f acc = {0.f, 0.f, 0.f, 0.f, 0.f, 0.f, 0.f, 0.f};
#pragma unroll
      for (int ks = 0; ks < 2; ++ks) {
        v16h b = load_b_swz(w2s, 128, ks, c * 16);
        acc = wmma16(a[ks], b, acc);
      }
      store_tile_f16(a2, 128, row0, c * 16, acc, b2, true);
    }
  }
  // ---- layer 3: 128x128 @ 128x256, fused relu + max over points ----
  {
    v16h a[4];
#pragma unroll
    for (int ks = 0; ks < 4; ++ks) a[ks] = load_a_frag(a2, 128, row0, ks * 32);
    const int nloc = w >> 2;  // waves 0-3 -> node0, 4-7 -> node1
    for (int c = 0; c < 16; ++c) {
      v8f acc = {0.f, 0.f, 0.f, 0.f, 0.f, 0.f, 0.f, 0.f};
#pragma unroll
      for (int ks = 0; ks < 4; ++ks) {
        v16h b = load_b_swz(w3s, 256, ks, c * 16);
        acc = wmma16(a[ks], b, acc);
      }
      int col = c * 16 + (lane & 15);
      float bv = b3[col];
      float mx = -3.402823466e+38f;
#pragma unroll
      for (int j = 0; j < 8; ++j) {
        float v = acc[j] + bv;
        v = fmaxf(v, 0.0f);
        mx = fmaxf(mx, v);
      }
      mx = fmaxf(mx, __shfl_xor(mx, 16, 32));  // combine row halves
      if (lane < 16) atomicMaxF(&featbuf[nloc * 256 + col], mx);
    }
  }
  __syncthreads();
  feat[node0 * 256 + t] = featbuf[t];
  feat[node0 * 256 + 256 + t] = featbuf[256 + t];
}

// ---------------------------------------------------------------------------
// Generic WMMA GEMM: C = act(A(f32, M x Kpad, fully K-padded) @ Bswz [+bias]).
// Block tile 128x64, 8 waves, K chunk 32.  B is pre-swizzled (Npad mult 64).
// ---------------------------------------------------------------------------
__global__ __launch_bounds__(256) void gemm_kernel(
    const float* __restrict__ A, const _Float16* __restrict__ Bsw,
    const float* __restrict__ bias, float* __restrict__ C, int M, int Kpad,
    int N, int npad, int do_relu) {
  __shared__ alignas(16) _Float16 As[128 * 32];
  __shared__ alignas(16) _Float16 Bs[32 * 64];  // swizzled, npad_lds = 64
  const int t = threadIdx.x, w = t >> 5, lane = t & 31;
  const int rb = blockIdx.x * 128, cb = blockIdx.y * 64;

  v8f acc[4];
  const v8f vzero = {0.f, 0.f, 0.f, 0.f, 0.f, 0.f, 0.f, 0.f};
#pragma unroll
  for (int c = 0; c < 4; ++c) acc[c] = vzero;

  // A staging coords: uniform per-thread row, contiguous 16 floats
  int ar = t >> 1, ac0 = (t & 1) * 16;
  int grow = rb + ar;
  if (grow >= M) grow = M - 1;  // clamp; garbage rows discarded at store
  // B staging coords: one v8h per thread of the 4KB chunk
  int bi = t * 8;
  int bhb = bi >> 10;            // 0/1 : K half-block
  int brem = bi & 1023;
  int bnl = brem >> 4;           // local col block
  int bj0 = brem & 15;           // 0 or 8

  const int KC = Kpad >> 5;
  for (int kc = 0; kc < KC; ++kc) {
    __syncthreads();
    {  // stage A (f32 -> f16), vectorizable: no per-element guards
      const float* ap = A + (size_t)grow * Kpad + kc * 32 + ac0;
      _Float16* dp = As + ar * 32 + ac0;
#pragma unroll
      for (int i = 0; i < 16; ++i) dp[i] = (_Float16)ap[i];
    }
    {  // stage B: contiguous copy from swizzled global
      const _Float16* sp =
          Bsw + ((size_t)((kc << 1) + bhb) * npad + cb + bnl) * 16 + bj0;
      *(v8h*)(Bs + ((bhb << 6) + bnl) * 16 + bj0) = *(const v8h*)sp;
    }
    __syncthreads();
    v16h a = load_a_frag(As, 32, w * 16, 0);
#pragma unroll
    for (int c = 0; c < 4; ++c) {
      v16h b = load_b_swz(Bs, 64, 0, c * 16);
      acc[c] = wmma16(a, b, acc[c]);
    }
  }
  // store
#pragma unroll
  for (int c = 0; c < 4; ++c) {
    int col = cb + c * 16 + (lane & 15);
    int rbase = rb + w * 16 + ((lane >> 4) << 3);
#pragma unroll
    for (int j = 0; j < 8; ++j) {
      int row = rbase + j;
      if (row < M && col < N) {
        float v = acc[c][j] + (bias ? bias[col] : 0.0f);
        if (do_relu) v = fmaxf(v, 0.0f);
        C[(size_t)row * N + col] = v;
      }
    }
  }
}

// ---------------------------------------------------------------------------
// Utility kernels
// ---------------------------------------------------------------------------
// Swizzle f32 KxN row-major -> f16 WMMA-B layout, zero-padded to Kpad x Npad.
__global__ void swizzle_b_kernel(const float* __restrict__ W,
                                 _Float16* __restrict__ out, int K, int N,
                                 int Kpad, int Npad) {
  int idx = blockIdx.x * blockDim.x + threadIdx.x;
  if (idx >= Kpad * Npad) return;
  int j = idx & 15;
  int t2 = idx >> 4;
  int n = t2 % Npad;
  int r = t2 / Npad;                       // kchunk*2 + hb
  int k = (r >> 1) * 32 + (r & 1) * 16 + j;
  float v = (k < K && n < N) ? W[(size_t)k * N + n] : 0.0f;
  out[idx] = (_Float16)v;
}

__global__ void fill_kernel(float* p, float v, int n) {
  int i = blockIdx.x * blockDim.x + threadIdx.x;
  if (i < n) p[i] = v;
}

// asrc/adst for GAT1 (8 heads x 10 dims): one thread per (node, head)
__global__ void attn1_kernel(const float* __restrict__ h,
                             const float* __restrict__ a_src,
                             const float* __restrict__ a_dst,
                             float* __restrict__ as, float* __restrict__ ad) {
  int idx = blockIdx.x * blockDim.x + threadIdx.x;
  if (idx >= N_NODES * 8) return;
  int hd = idx & 7;
  const float* hp = h + (idx >> 3) * 80 + hd * 10;
  float s = 0.f, d = 0.f;
#pragma unroll
  for (int k = 0; k < 10; ++k) {
    s += hp[k] * a_src[hd * 10 + k];
    d += hp[k] * a_dst[hd * 10 + k];
  }
  as[idx] = s;
  ad[idx] = d;
}

// asrc/adst for GAT2 (1 head x 512): one thread per node
__global__ void attn2_kernel(const float* __restrict__ h,
                             const float* __restrict__ a_src,
                             const float* __restrict__ a_dst,
                             float* __restrict__ as, float* __restrict__ ad) {
  int n = blockIdx.x * blockDim.x + threadIdx.x;
  if (n >= N_NODES) return;
  const float* hp = h + (size_t)n * 512;
  float s = 0.f, d = 0.f;
  for (int k = 0; k < 512; ++k) {
    s += hp[k] * a_src[k];
    d += hp[k] * a_dst[k];
  }
  as[n] = s;
  ad[n] = d;
}

// phase 1: e = leaky_relu(asrc[src]+adst[dst]); segment max over dst
__global__ void edge_max_kernel(const int* __restrict__ ei,
                                const float* __restrict__ asrc,
                                const float* __restrict__ adst,
                                float* __restrict__ ebuf, float* __restrict__ m,
                                int H) {
  int idx = blockIdx.x * blockDim.x + threadIdx.x;
  if (idx >= E_TOT * H) return;
  int e = idx / H, h = idx - e * H;
  int src, dst;
  if (e < N_EDGES) { src = ei[e]; dst = ei[N_EDGES + e]; }
  else             { src = e - N_EDGES; dst = src; }
  float v = asrc[src * H + h] + adst[dst * H + h];
  v = v > 0.f ? v : NEG_SLOPE * v;
  ebuf[idx] = v;
  atomicMaxF(&m[dst * H + h], v);
}

// phase 2: e = exp(e - m[dst]); segment sum over dst
__global__ void edge_exp_kernel(const int* __restrict__ ei,
                                float* __restrict__ ebuf,
                                const float* __restrict__ m,
                                float* __restrict__ s, int H) {
  int idx = blockIdx.x * blockDim.x + threadIdx.x;
  if (idx >= E_TOT * H) return;
  int e = idx / H, h = idx - e * H;
  int dst = (e < N_EDGES) ? ei[N_EDGES + e] : (e - N_EDGES);
  float ee = expf(ebuf[idx] - m[dst * H + h]);
  ebuf[idx] = ee;
  atomicAdd(&s[dst * H + h], ee);
}

// phase 3: out[dst] += h[src] * alpha  (one thread per (edge, feature))
// hin has row stride F = H*D; out has row stride OS (>= F, zero padded)
__global__ void edge_aggr_kernel(const int* __restrict__ ei,
                                 const float* __restrict__ ebuf,
                                 const float* __restrict__ s,
                                 const float* __restrict__ hin,
                                 float* __restrict__ out, int H, int D,
                                 int OS) {
  int idx = blockIdx.x * blockDim.x + threadIdx.x;
  int F = H * D;
  if (idx >= E_TOT * F) return;
  int e = idx / F, r = idx - e * F;
  int h = r / D;
  int src, dst;
  if (e < N_EDGES) { src = ei[e]; dst = ei[N_EDGES + e]; }
  else             { src = e - N_EDGES; dst = src; }
  float alpha = ebuf[e * H + h] / (s[dst * H + h] + GAT_EPS);
  atomicAdd(&out[(size_t)dst * OS + r], hin[(size_t)src * F + r] * alpha);
}

// in-place: p[row, col] = act(p[row, col] + bias[col]); row stride = stride
__global__ void bias_act_kernel(float* __restrict__ p,
                                const float* __restrict__ bias, int ncols,
                                int stride, int total, int do_relu) {
  int i = blockIdx.x * blockDim.x + threadIdx.x;
  if (i >= total) return;
  int row = i / ncols, col = i - row * ncols;
  float v = p[(size_t)row * stride + col] + bias[col];
  if (do_relu) v = fmaxf(v, 0.0f);
  p[(size_t)row * stride + col] = v;
}

// final: out[t] = (g[n0] * g[n1]) @ Wf + bf ; one wave (32 lanes) per edge
__global__ __launch_bounds__(256) void final_kernel(
    const float* __restrict__ g, const int* __restrict__ ei,
    const int* __restrict__ tid, const float* __restrict__ Wf,
    const float* __restrict__ bf, float* __restrict__ out) {
  int wv = ((blockIdx.x * blockDim.x + threadIdx.x) >> 5);
  int lane = threadIdx.x & 31;
  if (wv >= N_TRAIN) return;
  int id = tid[wv];
  int n0 = ei[id], n1 = ei[N_EDGES + id];
  const float* g0 = g + (size_t)n0 * 512;
  const float* g1 = g + (size_t)n1 * 512;
  float acc[7] = {0.f, 0.f, 0.f, 0.f, 0.f, 0.f, 0.f};
  for (int d = lane; d < 512; d += 32) {
    float f = g0[d] * g1[d];
#pragma unroll
    for (int c = 0; c < 7; ++c) acc[c] += f * Wf[d * 7 + c];
  }
#pragma unroll
  for (int c = 0; c < 7; ++c) {
#pragma unroll
    for (int off = 16; off > 0; off >>= 1)
      acc[c] += __shfl_xor(acc[c], off, 32);
  }
  if (lane == 0) {
#pragma unroll
    for (int c = 0; c < 7; ++c) out[wv * 7 + c] = acc[c] + bf[c];
  }
}

// ---------------------------------------------------------------------------
// Host orchestration
// ---------------------------------------------------------------------------
static inline int cdiv(long a, long b) { return (int)((a + b - 1) / b); }

extern "C" void kernel_launch(void* const* d_in, const int* in_sizes, int n_in,
                              void* d_out, int out_size, void* d_ws,
                              size_t ws_size, hipStream_t stream) {
  (void)in_sizes; (void)n_in; (void)out_size; (void)ws_size;
  const float* x    = (const float*)d_in[0];
  const int*   ei   = (const int*)d_in[1];
  const int*   tid  = (const int*)d_in[2];
  const float* W1   = (const float*)d_in[3];
  const float* b1   = (const float*)d_in[4];
  const float* W2   = (const float*)d_in[5];
  const float* b2   = (const float*)d_in[6];
  const float* W3   = (const float*)d_in[7];
  const float* b3   = (const float*)d_in[8];
  const float* Wg1  = (const float*)d_in[9];
  const float* as1w = (const float*)d_in[10];
  const float* ad1w = (const float*)d_in[11];
  const float* bg1  = (const float*)d_in[12];
  const float* Wg2  = (const float*)d_in[13];
  const float* as2w = (const float*)d_in[14];
  const float* ad2w = (const float*)d_in[15];
  const float* bg2  = (const float*)d_in[16];
  const float* Wl1  = (const float*)d_in[17];
  const float* bl1  = (const float*)d_in[18];
  const float* Wl2  = (const float*)d_in[19];
  const float* bl2  = (const float*)d_in[20];
  const float* Wf   = (const float*)d_in[21];
  const float* bf   = (const float*)d_in[22];
  float* out = (float*)d_out;

  char* ws = (char*)d_ws;
  size_t off = 0;
  auto alloc = [&](size_t bytes) -> char* {
    char* p = ws + off;
    off = (off + bytes + 255) & ~(size_t)255;
    return p;
  };
  // swizzled f16 weights (Kpad x Npad)
  _Float16* w1h  = (_Float16*)alloc((size_t)32 * 64 * 2);
  _Float16* w2h  = (_Float16*)alloc((size_t)64 * 128 * 2);
  _Float16* w3h  = (_Float16*)alloc((size_t)128 * 256 * 2);
  _Float16* wg1h = (_Float16*)alloc((size_t)256 * 128 * 2);
  _Float16* wg2h = (_Float16*)alloc((size_t)96 * 512 * 2);
  _Float16* wl1h = (_Float16*)alloc((size_t)512 * 512 * 2);
  _Float16* wl2h = (_Float16*)alloc((size_t)512 * 512 * 2);
  float* feat = (float*)alloc((size_t)N_NODES * 256 * 4);
  float* h1g  = (float*)alloc((size_t)N_NODES * 80 * 4);
  float* as1  = (float*)alloc((size_t)N_NODES * 8 * 4);
  float* ad1  = (float*)alloc((size_t)N_NODES * 8 * 4);
  float* m1   = (float*)alloc((size_t)N_NODES * 8 * 4);
  float* s1   = (float*)alloc((size_t)N_NODES * 8 * 4);
  float* e1   = (float*)alloc((size_t)E_TOT * 8 * 4);
  float* gat1 = (float*)alloc((size_t)N_NODES * 96 * 4);  // K-padded 80->96
  float* h2g  = (float*)alloc((size_t)N_NODES * 512 * 4);
  float* as2  = (float*)alloc((size_t)N_NODES * 4);
  float* ad2  = (float*)alloc((size_t)N_NODES * 4);
  float* m2   = (float*)alloc((size_t)N_NODES * 4);
  float* s2   = (float*)alloc((size_t)N_NODES * 4);
  float* e2   = (float*)alloc((size_t)E_TOT * 4);
  float* gat2 = (float*)alloc((size_t)N_NODES * 512 * 4);
  float* l1   = (float*)alloc((size_t)N_NODES * 512 * 4);
  float* l2   = (float*)alloc((size_t)N_NODES * 512 * 4);

  const int T = 256;
  // weight swizzle+convert (zero-padded)
  swizzle_b_kernel<<<cdiv(32 * 64, T), T, 0, stream>>>(W1, w1h, 16, 64, 32, 64);
  swizzle_b_kernel<<<cdiv(64 * 128, T), T, 0, stream>>>(W2, w2h, 64, 128, 64,
                                                        128);
  swizzle_b_kernel<<<cdiv(128 * 256, T), T, 0, stream>>>(W3, w3h, 128, 256, 128,
                                                         256);
  swizzle_b_kernel<<<cdiv(256 * 128, T), T, 0, stream>>>(Wg1, wg1h, 256, 80,
                                                         256, 128);
  swizzle_b_kernel<<<cdiv(96 * 512, T), T, 0, stream>>>(Wg2, wg2h, 80, 512, 96,
                                                        512);
  swizzle_b_kernel<<<cdiv(512 * 512, T), T, 0, stream>>>(Wl1, wl1h, 512, 512,
                                                         512, 512);
  swizzle_b_kernel<<<cdiv(512 * 512, T), T, 0, stream>>>(Wl2, wl2h, 512, 512,
                                                         512, 512);

  // fused point MLP + max-pool  -> feat (20000 x 256)
  point_mlp_kernel<<<N_NODES / 2, T, 0, stream>>>(x, w1h, w2h, w3h, b1, b2, b3,
                                                  feat);

  // ---- GAT layer 1 ----
  gemm_kernel<<<dim3(cdiv(N_NODES, 128), cdiv(80, 64)), T, 0, stream>>>(
      feat, wg1h, nullptr, h1g, N_NODES, 256, 80, 128, 0);
  attn1_kernel<<<cdiv(N_NODES * 8, T), T, 0, stream>>>(h1g, as1w, ad1w, as1,
                                                       ad1);
  fill_kernel<<<cdiv(N_NODES * 8, T), T, 0, stream>>>(m1, -3.402823466e+38f,
                                                      N_NODES * 8);
  fill_kernel<<<cdiv(N_NODES * 8, T), T, 0, stream>>>(s1, 0.0f, N_NODES * 8);
  fill_kernel<<<cdiv(N_NODES * 96, T), T, 0, stream>>>(gat1, 0.0f,
                                                       N_NODES * 96);
  edge_max_kernel<<<cdiv((long)E_TOT * 8, T), T, 0, stream>>>(ei, as1, ad1, e1,
                                                              m1, 8);
  edge_exp_kernel<<<cdiv((long)E_TOT * 8, T), T, 0, stream>>>(ei, e1, m1, s1,
                                                              8);
  edge_aggr_kernel<<<cdiv((long)E_TOT * 80, T), T, 0, stream>>>(
      ei, e1, s1, h1g, gat1, 8, 10, 96);
  bias_act_kernel<<<cdiv(N_NODES * 80, T), T, 0, stream>>>(
      gat1, bg1, 80, 96, N_NODES * 80, 1);

  // ---- GAT layer 2 ----
  gemm_kernel<<<dim3(cdiv(N_NODES, 128), cdiv(512, 64)), T, 0, stream>>>(
      gat1, wg2h, nullptr, h2g, N_NODES, 96, 512, 512, 0);
  attn2_kernel<<<cdiv(N_NODES, T), T, 0, stream>>>(h2g, as2w, ad2w, as2, ad2);
  fill_kernel<<<cdiv(N_NODES, T), T, 0, stream>>>(m2, -3.402823466e+38f,
                                                  N_NODES);
  fill_kernel<<<cdiv(N_NODES, T), T, 0, stream>>>(s2, 0.0f, N_NODES);
  fill_kernel<<<cdiv(N_NODES * 512, T), T, 0, stream>>>(gat2, 0.0f,
                                                        N_NODES * 512);
  edge_max_kernel<<<cdiv(E_TOT, T), T, 0, stream>>>(ei, as2, ad2, e2, m2, 1);
  edge_exp_kernel<<<cdiv(E_TOT, T), T, 0, stream>>>(ei, e2, m2, s2, 1);
  edge_aggr_kernel<<<cdiv((long)E_TOT * 512, T), T, 0, stream>>>(
      ei, e2, s2, h2g, gat2, 1, 512, 512);
  bias_act_kernel<<<cdiv(N_NODES * 512, T), T, 0, stream>>>(
      gat2, bg2, 512, 512, N_NODES * 512, 0);

  // ---- MLP head ----
  gemm_kernel<<<dim3(cdiv(N_NODES, 128), cdiv(512, 64)), T, 0, stream>>>(
      gat2, wl1h, bl1, l1, N_NODES, 512, 512, 512, 1);
  gemm_kernel<<<dim3(cdiv(N_NODES, 128), cdiv(512, 64)), T, 0, stream>>>(
      l1, wl2h, bl2, l2, N_NODES, 512, 512, 512, 0);

  // ---- edge fusion + classifier ----
  final_kernel<<<cdiv((long)N_TRAIN * 32, T), T, 0, stream>>>(l2, ei, tid, Wf,
                                                              bf, out);
}
